// SDCN_42185168781984
// MI455X (gfx1250) — matrix operand order repaired and checked by reading, not compiled, not run on hardware
//
#include <hip/hip_runtime.h>
#include <hip/hip_bf16.h>

// ---------------------------------------------------------------------------
// Types for CDNA5 WMMA (wave32)
// ---------------------------------------------------------------------------
typedef __attribute__((ext_vector_type(16))) __bf16 v16bf;
typedef __attribute__((ext_vector_type(8)))  __bf16 v8bf;
typedef __attribute__((ext_vector_type(8)))  float  v8f;

union V16 { v16bf v; v8bf h[2]; };

#define BM 128
#define BN 64
#define BK 64
#define LDS_STRIDE 72   // bf16 elems; 144B row stride -> conflict-free b128 reads

// Low 32 bits of a generic pointer to LDS == LDS byte offset (aperture is in
// the high half per CDNA5 flat addressing), which is what the async-to-LDS
// VDST operand wants.
__device__ __forceinline__ unsigned lds_off(const void* p) {
  return (unsigned)(unsigned long long)p;
}
// CDNA5 async DMA: global -> LDS, 16B per active lane, tracked by ASYNCcnt.
__device__ __forceinline__ void async_b128(unsigned dst_lds, const void* src) {
  asm volatile("global_load_async_to_lds_b128 %0, %1, off"
               :: "v"(dst_lds), "v"(src) : "memory");
}
__device__ __forceinline__ void wait_async0() {
  asm volatile("s_wait_asynccnt 0" ::: "memory");
}

// ---------------------------------------------------------------------------
// WMMA bf16 GEMM:  C[M,Nout] = A[M,KP] @ Bt[Nout,KP]^T (+bias) (+activation)
// A: bf16 row-major, ld = KP (multiple of 64, zero-padded)
// Bt: bf16 weight pre-transposed to [Nout][KP]
// act: 0=none, 1=clip(exp), 2=clip(softplus), 3=sigmoid
// Double-buffered LDS, async-to-LDS staging, 8 WMMA per K step per wave.
// ---------------------------------------------------------------------------
__global__ __launch_bounds__(256)
void k_gemm_bf16(const __bf16* __restrict__ A, const __bf16* __restrict__ Bt,
                 const float* __restrict__ bias, float* __restrict__ C,
                 int M, int Nout, int KP, int act) {
  __shared__ __bf16 As[2][BM * LDS_STRIDE];
  __shared__ __bf16 Bs[2][BN * LDS_STRIDE];

  const int tid  = threadIdx.x;
  const int lane = tid & 31;
  const int wave = tid >> 5;      // 0..7
  const int wm   = wave >> 1;     // 0..3
  const int wn   = wave & 1;      // 0..1
  const int m_blk = blockIdx.y * BM;
  const int n_blk = blockIdx.x * BN;
  const int row  = lane & 15;
  const int half = lane >> 4;

  // staging: A tile 128x64 bf16 (128B/row): thread -> (row, 32-elem chunk)
  const int ar = tid >> 1, ak = (tid & 1) * 32;
  // staging: B tile 64x64 bf16: thread -> (row, 16-elem chunk)
  const int br = tid >> 2, bk = (tid & 3) * 16;
  const int gr = m_blk + ar;
  const int gn = n_blk + br;
  const __bf16* aptr = A  + (size_t)gr * KP + ak;
  const __bf16* bptr = Bt + (size_t)gn * KP + bk;

  v8f acc[2][2] = {};

  auto stage = [&](int k0, int buf) {
    __bf16* adst = &As[buf][ar * LDS_STRIDE + ak];
    __bf16* bdst = &Bs[buf][br * LDS_STRIDE + bk];
    if (gr < M) {
      const __bf16* s = aptr + k0;
      async_b128(lds_off(adst +  0), s +  0);
      async_b128(lds_off(adst +  8), s +  8);
      async_b128(lds_off(adst + 16), s + 16);
      async_b128(lds_off(adst + 24), s + 24);
      if (k0 + BK < KP) __builtin_prefetch(s + BK, 0, 3);
    } else {
      v8bf z = {};
      *(v8bf*)(adst + 0)  = z; *(v8bf*)(adst + 8)  = z;
      *(v8bf*)(adst + 16) = z; *(v8bf*)(adst + 24) = z;
    }
    if (gn < Nout) {
      const __bf16* s = bptr + k0;
      async_b128(lds_off(bdst + 0), s + 0);
      async_b128(lds_off(bdst + 8), s + 8);
      if (k0 + BK < KP) __builtin_prefetch(s + BK, 0, 3);
    } else {
      v8bf z = {};
      *(v8bf*)(bdst + 0) = z; *(v8bf*)(bdst + 8) = z;
    }
  };

  stage(0, 0);
  int buf = 0;
  for (int k0 = 0; k0 < KP; k0 += BK) {
    wait_async0();       // my async fills done
    __syncthreads();     // everyone's fills visible, prior reads retired
    if (k0 + BK < KP) stage(k0 + BK, buf ^ 1);

#pragma unroll
    for (int kk = 0; kk < 2; ++kk) {
      V16 afr[2], bfr[2];
      // A 16x32 bf16 frag: lanes0-15 K0..7|K16..23, lanes16-31 +8 (ISA 7.12.2)
#pragma unroll
      for (int mi = 0; mi < 2; ++mi) {
        const __bf16* p = &As[buf][(wm * 32 + mi * 16 + row) * LDS_STRIDE + kk * 32];
        afr[mi].h[0] = *(const v8bf*)(p + half * 8);
        afr[mi].h[1] = *(const v8bf*)(p + 16 + half * 8);
      }
      // B 32x16 frag from transposed LDS tile: lane=col, K0..7|K16..23, lanes16-31 +8 (same as A)
#pragma unroll
      for (int ni = 0; ni < 2; ++ni) {
        const __bf16* p = &Bs[buf][(wn * 32 + ni * 16 + row) * LDS_STRIDE + kk * 32];
        bfr[ni].h[0] = *(const v8bf*)(p + half * 8);
        bfr[ni].h[1] = *(const v8bf*)(p + 16 + half * 8);
      }
#pragma unroll
      for (int mi = 0; mi < 2; ++mi)
#pragma unroll
        for (int ni = 0; ni < 2; ++ni)
          acc[mi][ni] = __builtin_amdgcn_wmma_f32_16x16x32_bf16(
              false, afr[mi].v, false, bfr[ni].v, (short)0, acc[mi][ni],
              false, false);
    }
    buf ^= 1;
  }

  // epilogue: C/D layout: lanes0-15 N=lane,M=r ; lanes16-31 N=lane-16,M=8+r
#pragma unroll
  for (int mi = 0; mi < 2; ++mi)
#pragma unroll
    for (int ni = 0; ni < 2; ++ni) {
      const int gcol = n_blk + wn * 32 + ni * 16 + (lane & 15);
      if (gcol >= Nout) continue;
      const float bv = bias ? bias[gcol] : 0.f;
#pragma unroll
      for (int r = 0; r < 8; ++r) {
        const int grow = m_blk + wm * 32 + mi * 16 + r + 8 * half;
        if (grow < M) {
          float v = acc[mi][ni][r] + bv;
          if (act == 1) {
            v = fminf(fmaxf(__expf(v), 1e-5f), 1e6f);
          } else if (act == 2) {
            float sp = (v > 20.f) ? v : log1pf(__expf(v));
            v = fminf(fmaxf(sp, 1e-4f), 1e4f);
          } else if (act == 3) {
            v = 1.f / (1.f + __expf(-v));
          }
          C[(size_t)grow * Nout + gcol] = v;
        }
      }
    }
}

// ---------------------------------------------------------------------------
// Elementwise / support kernels
// ---------------------------------------------------------------------------
__global__ void k_cvt_pad(const float* __restrict__ src, __bf16* __restrict__ dst,
                          int R, int C, int CP) {
  size_t idx = (size_t)blockIdx.x * blockDim.x + threadIdx.x;
  if (idx >= (size_t)R * CP) return;
  int r = (int)(idx / CP), c = (int)(idx % CP);
  float v = (c < C) ? src[(size_t)r * C + c] : 0.f;
  dst[idx] = (__bf16)v;
}

// W[K][O] f32 -> Wt[O][KP] bf16 (transposed, K zero-padded)
__global__ void k_wt_cvt(const float* __restrict__ W, __bf16* __restrict__ Wt,
                         int K, int O, int KP) {
  size_t idx = (size_t)blockIdx.x * blockDim.x + threadIdx.x;
  if (idx >= (size_t)O * KP) return;
  int o = (int)(idx / KP), k = (int)(idx % KP);
  Wt[idx] = (__bf16)((k < K) ? W[(size_t)k * O + o] : 0.f);
}

#define ROWCHUNK 1024
__global__ void k_colstats_part(const float* __restrict__ X, float* __restrict__ ps,
                                float* __restrict__ ps2, int R, int C) {
  int c = blockIdx.x * blockDim.x + threadIdx.x;
  if (c >= C) return;
  int ch = blockIdx.y;
  int r0 = ch * ROWCHUNK;
  int r1 = min(R, r0 + ROWCHUNK);
  float s = 0.f, s2 = 0.f;
  for (int r = r0; r < r1; ++r) {
    float v = X[(size_t)r * C + c];
    s += v; s2 += v * v;
  }
  ps[(size_t)ch * C + c]  = s;
  ps2[(size_t)ch * C + c] = s2;
}

__global__ void k_colstats_fin(const float* __restrict__ ps, const float* __restrict__ ps2,
                               float* __restrict__ mean, float* __restrict__ rstd,
                               int R, int C, int nch) {
  int c = blockIdx.x * blockDim.x + threadIdx.x;
  if (c >= C) return;
  float s = 0.f, s2 = 0.f;
  for (int ch = 0; ch < nch; ++ch) {
    s  += ps[(size_t)ch * C + c];
    s2 += ps2[(size_t)ch * C + c];
  }
  float m   = s / (float)R;
  float var = s2 / (float)R - m * m;   // biased var, matches jnp.var
  if (var < 0.f) var = 0.f;
  mean[c] = m;
  rstd[c] = rsqrtf(var + 1e-5f);
}

__global__ void k_bn_relu_cvt(const float* __restrict__ X, const float* __restrict__ mean,
                              const float* __restrict__ rstd, const float* __restrict__ g,
                              const float* __restrict__ b, __bf16* __restrict__ dst,
                              int R, int C, int CP) {
  size_t idx = (size_t)blockIdx.x * blockDim.x + threadIdx.x;
  if (idx >= (size_t)R * CP) return;
  int r = (int)(idx / CP), c = (int)(idx % CP);
  float v = 0.f;
  if (c < C) {
    v = (X[(size_t)r * C + c] - mean[c]) * rstd[c] * g[c] + b[c];
    v = fmaxf(v, 0.f);
  }
  dst[idx] = (__bf16)v;
}

// segment_sum over edges: out[dst[e], c] += ew[e] * S[src[e], c]
__global__ void k_scatter(const float* __restrict__ S, const int* __restrict__ src,
                          const int* __restrict__ dst, const float* __restrict__ ew,
                          float* __restrict__ out, int E, int C) {
  size_t idx = (size_t)blockIdx.x * blockDim.x + threadIdx.x;
  if (idx >= (size_t)E * C) return;
  int e = (int)(idx / C), c = (int)(idx % C);
  atomicAdd(&out[(size_t)dst[e] * C + c], ew[e] * S[(size_t)src[e] * C + c]);
}

// next = bf16( 0.5*relu(H) + 0.5*tra )   (SIGMA = 0.5)
__global__ void k_relu_mix_cvt(const float* __restrict__ H, const __bf16* __restrict__ T,
                               __bf16* __restrict__ dst, int R, int C, int CP) {
  size_t idx = (size_t)blockIdx.x * blockDim.x + threadIdx.x;
  if (idx >= (size_t)R * CP) return;
  int r = (int)(idx / CP), c = (int)(idx % CP);
  float v = 0.f;
  if (c < C) v = 0.5f * fmaxf(H[(size_t)r * C + c], 0.f) + 0.5f * (float)T[idx];
  dst[idx] = (__bf16)v;
}

__global__ void k_relu_mix_f32(const float* __restrict__ H, const float* __restrict__ Z,
                               float* __restrict__ out, size_t total) {
  size_t idx = (size_t)blockIdx.x * blockDim.x + threadIdx.x;
  if (idx >= total) return;
  out[idx] = 0.5f * fmaxf(H[idx], 0.f) + 0.5f * Z[idx];
}

// small-K dense layer (K <= 16): one block per row
__global__ __launch_bounds__(256)
void k_small_gemm(const float* __restrict__ A, const float* __restrict__ W,
                  const float* __restrict__ bias, float* __restrict__ out,
                  int K, int O) {
  int r = blockIdx.x;
  __shared__ float a[16];
  if (threadIdx.x < K) a[threadIdx.x] = A[(size_t)r * K + threadIdx.x];
  __syncthreads();
  for (int o = threadIdx.x; o < O; o += blockDim.x) {
    float s = bias ? bias[o] : 0.f;
    for (int k = 0; k < K; ++k) s += a[k] * W[(size_t)k * O + o];
    out[(size_t)r * O + o] = s;
  }
}

__global__ void k_softmax10(const float* __restrict__ X, float* __restrict__ out, int R) {
  int r = blockIdx.x * blockDim.x + threadIdx.x;
  if (r >= R) return;
  float mx = -1e30f;
  for (int c = 0; c < 10; ++c) mx = fmaxf(mx, X[(size_t)r * 10 + c]);
  float e[10], s = 0.f;
  for (int c = 0; c < 10; ++c) { e[c] = __expf(X[(size_t)r * 10 + c] - mx); s += e[c]; }
  float inv = 1.f / s;
  for (int c = 0; c < 10; ++c) out[(size_t)r * 10 + c] = e[c] * inv;
}

// student-t assignment, V = 1  =>  q_j = (1+dist_j)^-1, normalized
__global__ void k_q(const float* __restrict__ Z, const float* __restrict__ cluster,
                    float* __restrict__ out, int R) {
  int r = blockIdx.x * blockDim.x + threadIdx.x;
  if (r >= R) return;
  float z[10];
  for (int k = 0; k < 10; ++k) z[k] = Z[(size_t)r * 10 + k];
  float q[10], s = 0.f;
  for (int j = 0; j < 10; ++j) {
    float d = 0.f;
    for (int k = 0; k < 10; ++k) {
      float t = z[k] - cluster[(size_t)j * 10 + k];
      d += t * t;
    }
    q[j] = 1.f / (1.f + d);
    s += q[j];
  }
  float inv = 1.f / s;
  for (int j = 0; j < 10; ++j) out[(size_t)r * 10 + j] = q[j] * inv;
}

// ---------------------------------------------------------------------------
// Launcher
// ---------------------------------------------------------------------------
extern "C" void kernel_launch(void* const* d_in, const int* in_sizes, int n_in,
                              void* d_out, int out_size, void* d_ws, size_t ws_size,
                              hipStream_t stream) {
  const int D_IN = 2000;
  const int N = in_sizes[0] / D_IN;   // 20000
  const int E = in_sizes[1];          // 320000
  // padded K dims (multiple of BK=64)
  const int P2000 = 2048, P500 = 512;

  const float* x    = (const float*)d_in[0];
  const int*   esrc = (const int*)d_in[1];
  const int*   edst = (const int*)d_in[2];
  const float* ew   = (const float*)d_in[3];
  const float* enc1_W = (const float*)d_in[4];   const float* enc1_b = (const float*)d_in[5];
  const float* enc2_W = (const float*)d_in[6];   const float* enc2_b = (const float*)d_in[7];
  const float* enc3_W = (const float*)d_in[8];   const float* enc3_b = (const float*)d_in[9];
  const float* zl_W   = (const float*)d_in[10];  const float* zl_b   = (const float*)d_in[11];
  const float* dec1_W = (const float*)d_in[12];  const float* dec1_b = (const float*)d_in[13];
  const float* dec2_W = (const float*)d_in[14];  const float* dec2_b = (const float*)d_in[15];
  const float* dec3_W = (const float*)d_in[16];  const float* dec3_b = (const float*)d_in[17];
  const float* xbar_W = (const float*)d_in[18];  const float* xbar_b = (const float*)d_in[19];
  const float* mean_W = (const float*)d_in[20];  const float* mean_b = (const float*)d_in[21];
  const float* disp_W = (const float*)d_in[22];  const float* disp_b = (const float*)d_in[23];
  const float* piW    = (const float*)d_in[24];  const float* pi_b   = (const float*)d_in[25];
  const float* bn_g[6] = {(const float*)d_in[26], (const float*)d_in[28], (const float*)d_in[30],
                          (const float*)d_in[32], (const float*)d_in[34], (const float*)d_in[36]};
  const float* bn_b[6] = {(const float*)d_in[27], (const float*)d_in[29], (const float*)d_in[31],
                          (const float*)d_in[33], (const float*)d_in[35], (const float*)d_in[37]};
  const float* gnn1_W = (const float*)d_in[38];
  const float* gnn2_W = (const float*)d_in[39];
  const float* gnn3_W = (const float*)d_in[40];
  const float* gnn4_W = (const float*)d_in[41];
  const float* gnn5_W = (const float*)d_in[42];
  const float* cluster = (const float*)d_in[43];

  // d_out slots (return order)
  float* o_xbar = (float*)d_out;
  float* o_q    = o_xbar + (size_t)N * 2000;
  float* o_pred = o_q    + (size_t)N * 10;
  float* o_z    = o_pred + (size_t)N * 10;
  float* o_mean = o_z    + (size_t)N * 10;
  float* o_disp = o_mean + (size_t)N * 2000;
  float* o_pi   = o_disp + (size_t)N * 2000;

  // workspace carve
  char* w = (char*)d_ws;
  auto alloc = [&](size_t bytes) -> void* {
    void* p = (void*)w;
    w += (bytes + 255) & ~(size_t)255;
    return p;
  };
  const int nch = (N + ROWCHUNK - 1) / ROWCHUNK;
  __bf16* x_bf  = (__bf16*)alloc((size_t)N * P2000 * 2);
  __bf16* h1_bf = (__bf16*)alloc((size_t)N * P500 * 2);
  __bf16* h2_bf = (__bf16*)alloc((size_t)N * P500 * 2);
  __bf16* h3_bf = (__bf16*)alloc((size_t)N * P2000 * 2);
  __bf16* d1_bf = (__bf16*)alloc((size_t)N * P2000 * 2);
  __bf16* d2_bf = (__bf16*)alloc((size_t)N * P500 * 2);
  __bf16* d3_bf = (__bf16*)alloc((size_t)N * P500 * 2);
  __bf16* gx1   = (__bf16*)alloc((size_t)N * P500 * 2);
  __bf16* gx2   = (__bf16*)alloc((size_t)N * P500 * 2);
  __bf16* gx3   = (__bf16*)alloc((size_t)N * P2000 * 2);
  float*  T1    = (float*)alloc((size_t)N * 2000 * 4);   // GEMM output scratch
  float*  G     = (float*)alloc((size_t)N * 2000 * 4);   // scatter accumulator
  float*  mix5  = (float*)alloc((size_t)N * 10 * 4);
  float*  S5    = (float*)alloc((size_t)N * 10 * 4);
  float*  G5    = (float*)alloc((size_t)N * 10 * 4);
  float*  mbuf  = (float*)alloc(2048 * 4);
  float*  rbuf  = (float*)alloc(2048 * 4);
  float*  psum  = (float*)alloc((size_t)nch * 2000 * 4);
  float*  psum2 = (float*)alloc((size_t)nch * 2000 * 4);
  __bf16* Wt_enc1 = (__bf16*)alloc((size_t)500  * P2000 * 2);
  __bf16* Wt_enc2 = (__bf16*)alloc((size_t)500  * P500  * 2);
  __bf16* Wt_enc3 = (__bf16*)alloc((size_t)2000 * P500  * 2);
  __bf16* Wt_zl   = (__bf16*)alloc((size_t)10   * P2000 * 2);
  __bf16* Wt_dec2 = (__bf16*)alloc((size_t)500  * P2000 * 2);
  __bf16* Wt_dec3 = (__bf16*)alloc((size_t)500  * P500  * 2);
  __bf16* Wt_xbar = (__bf16*)alloc((size_t)2000 * P500  * 2);
  __bf16* Wt_mean = (__bf16*)alloc((size_t)2000 * P500  * 2);
  __bf16* Wt_disp = (__bf16*)alloc((size_t)2000 * P500  * 2);
  __bf16* Wt_pi   = (__bf16*)alloc((size_t)2000 * P500  * 2);
  __bf16* Wt_g1   = (__bf16*)alloc((size_t)500  * P2000 * 2);
  __bf16* Wt_g2   = (__bf16*)alloc((size_t)500  * P500  * 2);
  __bf16* Wt_g3   = (__bf16*)alloc((size_t)2000 * P500  * 2);
  __bf16* Wt_g4   = (__bf16*)alloc((size_t)10   * P2000 * 2);

  auto cdiv = [](size_t a, size_t b) -> unsigned { return (unsigned)((a + b - 1) / b); };
  auto ew_grid = [&](size_t total) { return dim3(cdiv(total, 256)); };

  auto gemm = [&](const __bf16* A, const __bf16* Bt, const float* bias, float* C,
                  int M, int Nout, int KP, int act) {
    dim3 grid((Nout + BN - 1) / BN, (M + BM - 1) / BM);
    k_gemm_bf16<<<grid, 256, 0, stream>>>(A, Bt, bias, C, M, Nout, KP, act);
  };
  auto wtc = [&](const float* W, __bf16* Wt, int K, int O, int KP) {
    k_wt_cvt<<<ew_grid((size_t)O * KP), 256, 0, stream>>>(W, Wt, K, O, KP);
  };
  auto stats = [&](const float* X, int C) {
    dim3 g(cdiv(C, 256), nch);
    k_colstats_part<<<g, 256, 0, stream>>>(X, psum, psum2, N, C);
    k_colstats_fin<<<cdiv(C, 256), 256, 0, stream>>>(psum, psum2, mbuf, rbuf, N, C, nch);
  };
  auto bnrelu = [&](const float* X, int bn, __bf16* dst, int C, int CP) {
    k_bn_relu_cvt<<<ew_grid((size_t)N * CP), 256, 0, stream>>>(
        X, mbuf, rbuf, bn_g[bn], bn_b[bn], dst, N, C, CP);
  };
  auto scatter = [&](const float* S, float* out, int C) {
    hipMemsetAsync(out, 0, (size_t)N * C * 4, stream);
    k_scatter<<<ew_grid((size_t)E * C), 256, 0, stream>>>(S, esrc, edst, ew, out, E, C);
  };

  // --- stage inputs / weights as bf16 ---
  k_cvt_pad<<<ew_grid((size_t)N * P2000), 256, 0, stream>>>(x, x_bf, N, D_IN, P2000);
  wtc(enc1_W, Wt_enc1, 2000, 500, P2000);
  wtc(enc2_W, Wt_enc2, 500, 500, P500);
  wtc(enc3_W, Wt_enc3, 500, 2000, P500);
  wtc(zl_W,   Wt_zl,   2000, 10, P2000);
  wtc(dec2_W, Wt_dec2, 2000, 500, P2000);
  wtc(dec3_W, Wt_dec3, 500, 500, P500);
  wtc(xbar_W, Wt_xbar, 500, 2000, P500);
  wtc(mean_W, Wt_mean, 500, 2000, P500);
  wtc(disp_W, Wt_disp, 500, 2000, P500);
  wtc(piW,    Wt_pi,   500, 2000, P500);
  wtc(gnn1_W, Wt_g1,   2000, 500, P2000);
  wtc(gnn2_W, Wt_g2,   500, 500, P500);
  wtc(gnn3_W, Wt_g3,   500, 2000, P500);
  wtc(gnn4_W, Wt_g4,   2000, 10, P2000);

  // --- autoencoder encoder ---
  gemm(x_bf, Wt_enc1, enc1_b, T1, N, 500, P2000, 0);
  stats(T1, 500);  bnrelu(T1, 0, h1_bf, 500, P500);
  gemm(h1_bf, Wt_enc2, enc2_b, T1, N, 500, P500, 0);
  stats(T1, 500);  bnrelu(T1, 1, h2_bf, 500, P500);
  gemm(h2_bf, Wt_enc3, enc3_b, T1, N, 2000, P500, 0);
  stats(T1, 2000); bnrelu(T1, 2, h3_bf, 2000, P2000);
  gemm(h3_bf, Wt_zl, zl_b, o_z, N, 10, P2000, 0);            // z -> output slot

  // --- decoder ---
  k_small_gemm<<<N, 256, 0, stream>>>(o_z, dec1_W, dec1_b, T1, 10, 2000);
  stats(T1, 2000); bnrelu(T1, 3, d1_bf, 2000, P2000);
  gemm(d1_bf, Wt_dec2, dec2_b, T1, N, 500, P2000, 0);
  stats(T1, 500);  bnrelu(T1, 4, d2_bf, 500, P500);
  gemm(d2_bf, Wt_dec3, dec3_b, T1, N, 500, P500, 0);
  stats(T1, 500);  bnrelu(T1, 5, d3_bf, 500, P500);

  // --- heads (fused epilogues) ---
  gemm(d3_bf, Wt_xbar, xbar_b, o_xbar, N, 2000, P500, 0);
  gemm(d3_bf, Wt_mean, mean_b, o_mean, N, 2000, P500, 1);
  gemm(d3_bf, Wt_disp, disp_b, o_disp, N, 2000, P500, 2);
  gemm(d3_bf, Wt_pi,   pi_b,   o_pi,   N, 2000, P500, 3);

  // --- GCN stack ---
  gemm(x_bf, Wt_g1, nullptr, T1, N, 500, P2000, 0);
  scatter(T1, G, 500);
  k_relu_mix_cvt<<<ew_grid((size_t)N * P500), 256, 0, stream>>>(G, h1_bf, gx1, N, 500, P500);

  gemm(gx1, Wt_g2, nullptr, T1, N, 500, P500, 0);
  scatter(T1, G, 500);
  k_relu_mix_cvt<<<ew_grid((size_t)N * P500), 256, 0, stream>>>(G, h2_bf, gx2, N, 500, P500);

  gemm(gx2, Wt_g3, nullptr, T1, N, 2000, P500, 0);
  scatter(T1, G, 2000);
  k_relu_mix_cvt<<<ew_grid((size_t)N * P2000), 256, 0, stream>>>(G, h3_bf, gx3, N, 2000, P2000);

  gemm(gx3, Wt_g4, nullptr, T1, N, 10, P2000, 0);
  scatter(T1, G, 10);   // reuse G's first N*10 region
  k_relu_mix_f32<<<ew_grid((size_t)N * 10), 256, 0, stream>>>(G, o_z, mix5, (size_t)N * 10);

  k_small_gemm<<<N, 256, 0, stream>>>(mix5, gnn5_W, nullptr, S5, 10, 10);
  scatter(S5, G5, 10);
  k_softmax10<<<cdiv(N, 256), 256, 0, stream>>>(G5, o_pred, N);

  // --- student-t q ---
  k_q<<<cdiv(N, 256), 256, 0, stream>>>(o_z, cluster, o_q, N);
}
